// xDeepFM_CIN_Layer_63763084477357
// MI455X (gfx1250) — compile-verified, matrix operand order checked by reading, not run on hardware
//
#include <hip/hip_runtime.h>

// CDNA5 (gfx1250) fused xDeepFM CIN: 3 layers on v_wmma_f32_16x16x32_f16.
// B=2048, NF=32, EMB(K)=64, O=128 per layer. Output (2048, 256) f32.
//
// Compute-bound (172 GFLOP vs ~21 MB): all matmul work on the matrix pipe.
// A-fragments are synthesized algebraically (per-row scalar x fixed 32-vec,
// 8x v_pk_mul_f16); W is pre-packed to B-fragment layout and streamed from
// the 192MB L2; inp is pre-packed to A-fragment layout and staged into LDS
// with GLOBAL_LOAD_ASYNC_TO_LDS_B128 (ASYNCcnt). Two k-positions per wave
// share every B fragment (halves W traffic, doubles WMMA density).

typedef __attribute__((ext_vector_type(16))) _Float16 v16h;
typedef __attribute__((ext_vector_type(8)))  float    v8f;

// ---------------------------------------------------------------------------
// Prep 1: pack W (f32, row-major O x IC) into f16 WMMA-B fragment layout.
// Per (j-chunk, n-tile): 32 lanes x 16 halves contiguous (1 KB).
// Assumed 16-bit 32x16 B layout (mirror of documented A layout, N<->M roles):
//   lane L (L&15 = n-in-tile, hi = L>>4) half h holds K = j*32 + z,
//   z = (h>>3)*16 + hi*8 + (h&7).
// ---------------------------------------------------------------------------
__global__ void pack_w_kernel(const float* __restrict__ W,
                              _Float16* __restrict__ out,
                              int IC, int nj) {
  int e = blockIdx.x * 256 + threadIdx.x;
  if (e >= nj * 4096) return;          // nj * 8 tiles * 32 lanes * 16 halves
  int h    = e & 15;
  int lane = (e >> 4) & 31;
  int t    = (e >> 9) & 7;
  int j    = e >> 12;
  int n    = t * 16 + (lane & 15);
  int hi   = lane >> 4;
  int z    = ((h >> 3) << 4) + hi * 8 + (h & 7);
  out[e] = (_Float16)W[(size_t)n * IC + j * 32 + z];
}

// ---------------------------------------------------------------------------
// Prep 2: pack inp (f32, (2048,32,64)) into f16 WMMA-A fragment layout:
// inpA_g[btile(128)][k(64)][slot(32)][half(16)], 1 KB per (btile,k).
// Makes the main kernel's LDS staging a pure async byte copy.
// ---------------------------------------------------------------------------
__global__ void pack_inp_kernel(const float* __restrict__ inp,
                                _Float16* __restrict__ out) {
  int e = blockIdx.x * 256 + threadIdx.x;   // 128*64*512 = 4194304 total
  int hf   = e & 15;
  int slot = (e >> 4) & 31;
  int k    = (e >> 9) & 63;
  int bt   = e >> 15;
  int m  = slot & 15, hi = slot >> 4;
  int z  = ((hf >> 3) << 4) + hi * 8 + (hf & 7);
  out[e] = (_Float16)inp[(((size_t)(bt * 16 + m)) * 32 + z) * 64 + k];
}

__device__ __forceinline__ v16h splat16(_Float16 x) {
  v16h v;
#pragma unroll
  for (int i = 0; i < 16; ++i) v[i] = x;
  return v;
}

// One CIN layer for a k-pair: every B fragment feeds two WMMAs.
template <int NJ, class FH0, class FH1>
__device__ __forceinline__ void layer_mm(const v16h base0, const v16h base1,
                                         FH0 hget0, FH1 hget1,
                                         const _Float16* __restrict__ wpk,
                                         int lane, v8f c0[8], v8f c1[8]) {
  _Float16 h0 = hget0(0), h1 = hget1(0);
#pragma unroll 1
  for (int j = 0; j < NJ; ++j) {
    v16h a0 = base0 * splat16(h0);          // 8x v_pk_mul_f16 each
    v16h a1 = base1 * splat16(h1);
    int jn = (j + 1 < NJ) ? j + 1 : j;      // pipeline next h scalar
    h0 = hget0(jn);
    h1 = hget1(jn);
    const _Float16* wb = wpk + (size_t)(j * 8) * 512 + lane * 16;
    __builtin_prefetch(wb + 8 * 512, 0, 3);
#pragma unroll
    for (int t = 0; t < 8; ++t) {
      v16h bf = *(const v16h*)(wb + t * 512);
      c0[t] = __builtin_amdgcn_wmma_f32_16x16x32_f16(
          false, a0, false, bf, (short)0, c0[t], false, false);
      c1[t] = __builtin_amdgcn_wmma_f32_16x16x32_f16(
          false, a1, false, bf, (short)0, c1[t], false, false);
    }
  }
}

__device__ __forceinline__ void store_h(const v8f c[8], _Float16* hrow,
                                        int nn, int hi) {
#pragma unroll
  for (int t = 0; t < 4; ++t)              // h_next = nxt[:, 0:64] as f16
#pragma unroll
    for (int r = 0; r < 8; ++r)
      hrow[(t * 16 + nn) * 16 + r + 8 * hi] = (_Float16)c[t][r];
}

template <int T0, int T1>
__device__ __forceinline__ void add_out(const v8f c[8], float (*oacc)[256],
                                        int cbase, int nn, int hi) {
#pragma unroll
  for (int t = T0; t < T1; ++t)
#pragma unroll
    for (int r = 0; r < 8; ++r)            // ds_add_f32 reduction over k/waves
      atomicAdd(&oacc[r + 8 * hi][cbase + (t - T0) * 16 + nn], c[t][r]);
}

// ---------------------------------------------------------------------------
// Fused 3-layer CIN. Grid: 128 blocks (16-row batch tiles) x 128 threads
// (4 waves). 4 phases of 16 k async-staged into LDS; each wave runs 2 k-pairs
// per phase through L0->L1->L2 entirely in registers.
// ---------------------------------------------------------------------------
__global__ __launch_bounds__(128) void cin_fused_kernel(
    const _Float16* __restrict__ inpA_g,
    const _Float16* __restrict__ wpk0, const _Float16* __restrict__ wpk1,
    const _Float16* __restrict__ wpk2,
    const float* __restrict__ b0, const float* __restrict__ b1,
    const float* __restrict__ b2, float* __restrict__ out) {
  __shared__ _Float16 inpA[16 * 512];        // 16 KB: phase's A-frag data
  __shared__ _Float16 hbuf[4][2][64][16];    // 16 KB: per-wave, per-pair h
  __shared__ float    outAcc[16][256];       // 16 KB: block output sums
  __shared__ float    biasL[3][128];         // 1.5 KB

  const int tid = threadIdx.x, lane = tid & 31, wv = tid >> 5;
  const int nn = lane & 15, hi = lane >> 4;

  for (int i = tid; i < 16 * 256; i += 128) ((float*)outAcc)[i] = 0.0f;
  biasL[0][tid] = b0[tid];
  biasL[1][tid] = b1[tid];
  biasL[2][tid] = b2[tid];

  const uint32_t lds0 = (uint32_t)(uintptr_t)(&inpA[0]);
  const uint64_t gbase =
      (uint64_t)(uintptr_t)(inpA_g + (size_t)blockIdx.x * 64 * 512);

#pragma unroll 1
  for (int ph = 0; ph < 4; ++ph) {
    __syncthreads();  // previous phase fully consumed before overwrite
    // Async stage 16 KB (16 k x 1 KB) of A-fragment data into LDS.
#pragma unroll
    for (int cpy = 0; cpy < 4; ++cpy) {
      uint32_t off  = (uint32_t)((tid + cpy * 128) * 32);
      uint32_t goff = (uint32_t)(ph * 16 * 1024) + off;
      asm volatile("global_load_async_to_lds_b128 %0, %1, %2"
                   :: "v"(lds0 + off), "v"(goff), "s"(gbase) : "memory");
      asm volatile("global_load_async_to_lds_b128 %0, %1, %2 offset:16"
                   :: "v"(lds0 + off), "v"(goff), "s"(gbase) : "memory");
    }
    asm volatile("s_wait_asynccnt 0x0" ::: "memory");
    __syncthreads();

#pragma unroll 1
    for (int pr = 0; pr < 2; ++pr) {
      const int kl0 = wv * 4 + pr * 2;       // k_local pair within phase
      const int kl1 = kl0 + 1;
      const v16h base0 = *(const v16h*)&inpA[kl0 * 512 + lane * 16];
      const v16h base1 = *(const v16h*)&inpA[kl1 * 512 + lane * 16];
      v8f c0[8], c1[8];

      // ---------------- Layer 0: h = inp (j = 0..31) ----------------
#pragma unroll
      for (int t = 0; t < 8; ++t) {
        float bv = biasL[0][t * 16 + nn];
#pragma unroll
        for (int r = 0; r < 8; ++r) { c0[t][r] = bv; c1[t][r] = bv; }
      }
      auto h00 = [&](int j) -> _Float16 {
        return inpA[kl0 * 512 + (nn + 16 * ((j >> 3) & 1)) * 16 +
                    (j & 7) + 8 * (j >> 4)];
      };
      auto h01 = [&](int j) -> _Float16 {
        return inpA[kl1 * 512 + (nn + 16 * ((j >> 3) & 1)) * 16 +
                    (j & 7) + 8 * (j >> 4)];
      };
      layer_mm<32>(base0, base1, h00, h01, wpk0, lane, c0, c1);
      store_h(c0, &hbuf[wv][0][0][0], nn, hi);
      store_h(c1, &hbuf[wv][1][0][0], nn, hi);
      add_out<4, 8>(c0, outAcc, 0, nn, hi);      // out channels 0..63
      add_out<4, 8>(c1, outAcc, 0, nn, hi);

      auto h10 = [&](int j) -> _Float16 { return hbuf[wv][0][j][nn]; };
      auto h11 = [&](int j) -> _Float16 { return hbuf[wv][1][j][nn]; };

      // ---------------- Layer 1: h = h1 (j = 0..63) ----------------
#pragma unroll
      for (int t = 0; t < 8; ++t) {
        float bv = biasL[1][t * 16 + nn];
#pragma unroll
        for (int r = 0; r < 8; ++r) { c0[t][r] = bv; c1[t][r] = bv; }
      }
      layer_mm<64>(base0, base1, h10, h11, wpk1, lane, c0, c1);
      store_h(c0, &hbuf[wv][0][0][0], nn, hi);
      store_h(c1, &hbuf[wv][1][0][0], nn, hi);
      add_out<4, 8>(c0, outAcc, 64, nn, hi);     // out channels 64..127
      add_out<4, 8>(c1, outAcc, 64, nn, hi);

      // ---------------- Layer 2: h = h2, all 128 channels out --------
#pragma unroll
      for (int t = 0; t < 8; ++t) {
        float bv = biasL[2][t * 16 + nn];
#pragma unroll
        for (int r = 0; r < 8; ++r) { c0[t][r] = bv; c1[t][r] = bv; }
      }
      layer_mm<64>(base0, base1, h10, h11, wpk2, lane, c0, c1);
      add_out<0, 8>(c0, outAcc, 128, nn, hi);    // out channels 128..255
      add_out<0, 8>(c1, outAcc, 128, nn, hi);
    }
  }

  __syncthreads();
  // Coalesced store: thread -> (row m = tid>>3, 32 channels at (tid&7)*32).
  {
    int m  = tid >> 3;
    int cb = (tid & 7) * 32;
    float* orow = out + ((size_t)blockIdx.x * 16 + m) * 256 + cb;
#pragma unroll
    for (int i = 0; i < 32; i += 4)
      *(float4*)(orow + i) = *(float4*)&outAcc[m][cb + i];
  }
}

// ---------------------------------------------------------------------------
extern "C" void kernel_launch(void* const* d_in, const int* in_sizes, int n_in,
                              void* d_out, int out_size, void* d_ws,
                              size_t ws_size, hipStream_t stream) {
  const float* inp = (const float*)d_in[0];
  const float* W0  = (const float*)d_in[1];
  const float* b0  = (const float*)d_in[2];
  const float* W1  = (const float*)d_in[3];
  const float* b1  = (const float*)d_in[4];
  const float* W2  = (const float*)d_in[5];
  const float* b2  = (const float*)d_in[6];

  // Workspace: packed f16 weights (1.25 MB) + packed f16 inp (8 MB).
  _Float16* wpk0   = (_Float16*)d_ws;
  _Float16* wpk1   = wpk0 + (size_t)32 * 4096;
  _Float16* wpk2   = wpk1 + (size_t)64 * 4096;
  _Float16* inpA_g = wpk2 + (size_t)64 * 4096;

  pack_w_kernel<<<(32 * 4096 + 255) / 256, 256, 0, stream>>>(W0, wpk0, 1024, 32);
  pack_w_kernel<<<(64 * 4096 + 255) / 256, 256, 0, stream>>>(W1, wpk1, 2048, 64);
  pack_w_kernel<<<(64 * 4096 + 255) / 256, 256, 0, stream>>>(W2, wpk2, 2048, 64);
  pack_inp_kernel<<<16384, 256, 0, stream>>>(inp, inpA_g);

  cin_fused_kernel<<<128, 128, 0, stream>>>(inpA_g, wpk0, wpk1, wpk2,
                                            b0, b1, b2, (float*)d_out);
}
